// MultiHeadAttention_53515292508706
// MI455X (gfx1250) — matrix operand order repaired and checked by reading, not compiled, not run on hardware
//
#include <hip/hip_runtime.h>

typedef __bf16 bf16;
typedef __attribute__((ext_vector_type(8)))  __bf16 v8bf;
typedef __attribute__((ext_vector_type(16))) __bf16 v16bf;
typedef __attribute__((ext_vector_type(8)))  float  v8f;

#define BATCH 4
#define NSEQ  2048
#define DIM   1024
#define HEADS 16
#define HD    64
#define MTOT  (BATCH * NSEQ)       // 8192
#define SCALE 0.125f               // 1/sqrt(64)

// ---------------------------------------------------------------------------
// WMMA helpers (bf16 inputs, fp32 accumulate) — ISA layouts from CDNA5 ISA
// ---------------------------------------------------------------------------
static __device__ __forceinline__ v8f wmma_bf16(v16bf a, v16bf b, v8f c) {
  return __builtin_amdgcn_wmma_f32_16x16x32_bf16(
      /*neg_a=*/false, a, /*neg_b=*/false, b,
      /*c_mod=*/(short)0, c, /*reuse_a=*/false, /*reuse_b=*/false);
}

static __device__ __forceinline__ v16bf make_frag(const bf16* p0, const bf16* p1) {
  v8bf lo = *(const v8bf*)p0;
  v8bf hi = *(const v8bf*)p1;
  v16bf r;
#pragma unroll
  for (int i = 0; i < 8; ++i) { r[i] = lo[i]; r[i + 8] = hi[i]; }
  return r;
}

// A-matrix 16x32 bf16 fragment. Row-major A, leading dim ld.
// lane<16: M=lane, K = {k..k+7, k+16..k+23}; lane>=16: M=lane-16, K = {k+8..k+15, k+24..k+31}
static __device__ __forceinline__ v16bf load_a(const bf16* A, int rowBase, int ld,
                                               int k, int lane) {
  int m    = rowBase + (lane & 15);
  int koff = k + ((lane >> 4) << 3);
  const bf16* p = A + (size_t)m * ld + koff;
  return make_frag(p, p + 16);
}

// B-matrix 32x16 bf16 fragment where column n of B is ROW (colBase+n) of storage Bm
// (leading dim ld). lane holds column n=lane&15, K = (lane<16 ? {k..k+7, k+16..k+23}
// : {k+8..k+15, k+24..k+31}), matching the A-fragment K order.
static __device__ __forceinline__ v16bf load_b(const bf16* Bm, int colBase, int ld,
                                               int k, int lane) {
  int n = colBase + (lane & 15);
  const bf16* p = Bm + (size_t)n * ld + k + ((lane >> 4) << 3);
  return make_frag(p, p + 16);
}

// 16-lane row reductions over the C/D layout (row m striped over lanes 0-15 / 16-31)
static __device__ __forceinline__ float rowmax16(float v) {
#pragma unroll
  for (int m = 1; m <= 8; m <<= 1) v = fmaxf(v, __shfl_xor(v, m, 32));
  return v;
}
static __device__ __forceinline__ float rowsum16(float v) {
#pragma unroll
  for (int m = 1; m <= 8; m <<= 1) v += __shfl_xor(v, m, 32);
  return v;
}

// ---------------------------------------------------------------------------
// 0) fp32 -> bf16 conversion
// ---------------------------------------------------------------------------
__global__ void cvt_bf16_kernel(const float* __restrict__ src, bf16* __restrict__ dst,
                                int n) {
  int i = blockIdx.x * blockDim.x + threadIdx.x;
  int stride = gridDim.x * blockDim.x;
  for (; i < n; i += stride) dst[i] = (bf16)src[i];
}

// ---------------------------------------------------------------------------
// 1) QKV projection: y[m,e] = sum_d x[m,d] * w_qkv[e,d]
//    Writes q,k as [B,H,N,hd] bf16, v transposed as [B,H,hd,N] bf16.
//    Wave computes a 32x64 tile (8 WMMA per k-step, ~22 FLOP/B, ~180 VGPRs);
//    block = 4 waves = 64x128.
// ---------------------------------------------------------------------------
static __device__ __forceinline__ void store_qkv_tile(v8f c, int mBase, int eBase,
                                                      int lane, bf16* qb, bf16* kb,
                                                      bf16* vtb) {
  int hi8 = (lane >> 4) << 3;
#pragma unroll
  for (int r = 0; r < 8; ++r) {
    int m = mBase + r + hi8;
    int e = eBase + (lane & 15);
    int seg = e >> 10;           // 0=q 1=k 2=v
    int ee  = e & 1023;
    int h   = ee >> 6;
    int d   = ee & 63;
    int b   = m >> 11;           // m / 2048
    int n   = m & 2047;
    bf16 v = (bf16)c[r];
    size_t bh = (size_t)(b * HEADS + h);
    if (seg == 0)      qb[(bh * NSEQ + n) * HD + d] = v;
    else if (seg == 1) kb[(bh * NSEQ + n) * HD + d] = v;
    else               vtb[(bh * HD + d) * NSEQ + n] = v;
  }
}

__global__ void __launch_bounds__(128, 1)
gemm_qkv_kernel(const bf16* __restrict__ X, const bf16* __restrict__ W,
                bf16* __restrict__ qb, bf16* __restrict__ kb,
                bf16* __restrict__ vtb) {
  const int lane = threadIdx.x & 31;
  const int wave = threadIdx.x >> 5;
  const int mB = blockIdx.y * 64 + (wave & 1) * 32;
  const int eB = blockIdx.x * 128 + (wave >> 1) * 64;

  v8f c[2][4];
#pragma unroll
  for (int i = 0; i < 2; ++i)
#pragma unroll
    for (int j = 0; j < 4; ++j) c[i][j] = {};

  for (int k = 0; k < DIM; k += 32) {
    if (k + 128 < DIM) {   // uniform; distance ~4 k-steps -> global_prefetch_b8
      __builtin_prefetch(X + (size_t)(mB + (lane & 31)) * DIM + k + 128, 0, 0);
      __builtin_prefetch(W + (size_t)(eB + (lane & 31)) * DIM + k + 128, 0, 0);
      __builtin_prefetch(W + (size_t)(eB + 32 + (lane & 31)) * DIM + k + 128, 0, 0);
    }
    v16bf a[2], b[4];
#pragma unroll
    for (int i = 0; i < 2; ++i) a[i] = load_a(X, mB + 16 * i, DIM, k, lane);
#pragma unroll
    for (int j = 0; j < 4; ++j) b[j] = load_b(W, eB + 16 * j, DIM, k, lane);
#pragma unroll
    for (int i = 0; i < 2; ++i)
#pragma unroll
      for (int j = 0; j < 4; ++j) c[i][j] = wmma_bf16(a[i], b[j], c[i][j]);
  }
#pragma unroll
  for (int i = 0; i < 2; ++i)
#pragma unroll
    for (int j = 0; j < 4; ++j)
      store_qkv_tile(c[i][j], mB + 16 * i, eB + 16 * j, lane, qb, kb, vtb);
}

// ---------------------------------------------------------------------------
// 2) Flash attention per (b,h). Wave owns 16 query rows; loop keys in blocks
//    of 64 (amortizes the 16-lane softmax reductions). S via 8 WMMA, online
//    softmax fp32, P staged bf16 through LDS to re-layout D->A fragment,
//    P@V via 8 WMMA (V pre-transposed so B-frags are contiguous).
// ---------------------------------------------------------------------------
__global__ void __launch_bounds__(128, 1)
attn_kernel(const bf16* __restrict__ Q, const bf16* __restrict__ K,
            const bf16* __restrict__ VT, bf16* __restrict__ AO) {
  const int lane = threadIdx.x & 31;
  const int wave = threadIdx.x >> 5;
  const int bh = blockIdx.y;                 // 0..B*H-1
  const int b = bh >> 4, h = bh & 15;
  const int rowBase = blockIdx.x * 64 + wave * 16;

  const bf16* q  = Q  + (size_t)bh * NSEQ * HD;
  const bf16* kk = K  + (size_t)bh * NSEQ * HD;
  const bf16* vt = VT + (size_t)bh * HD * NSEQ;

  __shared__ bf16 lds[4][16 * 64];
  bf16* pt = lds[wave];

  // Q fragments stay resident (16x64 = two 16x32 A frags)
  v16bf qa0 = load_a(q, rowBase, HD, 0, lane);
  v16bf qa1 = load_a(q, rowBase, HD, 32, lane);

  float mrow[8], lrow[8];
  v8f o[4] = {{}, {}, {}, {}};
#pragma unroll
  for (int r = 0; r < 8; ++r) { mrow[r] = -3.0e38f; lrow[r] = 0.f; }

  const int hi8 = (lane >> 4) << 3;

  for (int kb0 = 0; kb0 < NSEQ; kb0 += 64) {
    if (kb0 + 64 < NSEQ) {   // prefetch next K rows / V columns (streaming)
      __builtin_prefetch(kk + (size_t)(kb0 + 64 + (lane & 31)) * HD, 0, 0);
      __builtin_prefetch(kk + (size_t)(kb0 + 96 + (lane & 31)) * HD, 0, 0);
      __builtin_prefetch(vt + (size_t)(lane & 31) * NSEQ + kb0 + 64, 0, 0);
      __builtin_prefetch(vt + (size_t)(32 + (lane & 31)) * NSEQ + kb0 + 64, 0, 0);
    }
    // S = Q @ K^T for 64 keys (four 16x16 tiles), fp32
    v8f s[4] = {{}, {}, {}, {}};
#pragma unroll
    for (int t = 0; t < 4; ++t) {
      v16bf bk;
      bk = load_b(kk, kb0 + 16 * t, HD, 0,  lane); s[t] = wmma_bf16(qa0, bk, s[t]);
      bk = load_b(kk, kb0 + 16 * t, HD, 32, lane); s[t] = wmma_bf16(qa1, bk, s[t]);
    }
    // online softmax (per row r of this wave's 16 rows)
#pragma unroll
    for (int r = 0; r < 8; ++r) {
      float v0 = s[0][r] * SCALE, v1 = s[1][r] * SCALE;
      float v2 = s[2][r] * SCALE, v3 = s[3][r] * SCALE;
      float mnew = fmaxf(mrow[r],
                         rowmax16(fmaxf(fmaxf(v0, v1), fmaxf(v2, v3))));
      float fac = __expf(mrow[r] - mnew);
      mrow[r] = mnew;
      float e0 = __expf(v0 - mnew), e1 = __expf(v1 - mnew);
      float e2 = __expf(v2 - mnew), e3 = __expf(v3 - mnew);
      lrow[r] = lrow[r] * fac + rowsum16((e0 + e1) + (e2 + e3));
#pragma unroll
      for (int t = 0; t < 4; ++t) o[t][r] *= fac;
      // stage P (bf16) into LDS: row (r+hi8), key col (lane&15) + 16*t
      int rb = (r + hi8) * 64 + (lane & 15);
      pt[rb +  0] = (bf16)e0;
      pt[rb + 16] = (bf16)e1;
      pt[rb + 32] = (bf16)e2;
      pt[rb + 48] = (bf16)e3;
    }
    __syncthreads();
    v16bf pa0, pa1;
    {
      const bf16* p = pt + (lane & 15) * 64 + ((lane >> 4) << 3);
      pa0 = make_frag(p,      p + 16);   // keys kb0+0..31
      pa1 = make_frag(p + 32, p + 48);   // keys kb0+32..63
    }
    __syncthreads();
    // O += P @ V (V stored transposed: column d contiguous over keys)
#pragma unroll
    for (int t = 0; t < 4; ++t) {
      v16bf bv;
      bv = load_b(vt, 16 * t, NSEQ, kb0,      lane); o[t] = wmma_bf16(pa0, bv, o[t]);
      bv = load_b(vt, 16 * t, NSEQ, kb0 + 32, lane); o[t] = wmma_bf16(pa1, bv, o[t]);
    }
  }

  // normalize and write attn output bf16 [B,N,D] (feature = h*64 + d)
#pragma unroll
  for (int r = 0; r < 8; ++r) {
    float inv = 1.0f / lrow[r];
    int n = rowBase + r + hi8;
    size_t base = ((size_t)b * NSEQ + n) * DIM + h * HD;
    int d = lane & 15;
#pragma unroll
    for (int t = 0; t < 4; ++t) AO[base + 16 * t + d] = (bf16)(o[t][r] * inv);
  }
}

// ---------------------------------------------------------------------------
// 3) Output projection: out[m,e] = sum_d ao[m,d]*w_out[e,d] + b_out[e]  (fp32 out)
// ---------------------------------------------------------------------------
static __device__ __forceinline__ void store_out_tile(v8f c, int mBase, int eBase,
                                                      int lane,
                                                      const float* __restrict__ bias,
                                                      float* __restrict__ out) {
  int hi8 = (lane >> 4) << 3;
#pragma unroll
  for (int r = 0; r < 8; ++r) {
    int m = mBase + r + hi8;
    int e = eBase + (lane & 15);
    out[(size_t)m * DIM + e] = c[r] + bias[e];
  }
}

__global__ void __launch_bounds__(128, 1)
gemm_out_kernel(const bf16* __restrict__ A, const bf16* __restrict__ W,
                const float* __restrict__ bias, float* __restrict__ out) {
  const int lane = threadIdx.x & 31;
  const int wave = threadIdx.x >> 5;
  const int mB = blockIdx.y * 64 + (wave & 1) * 32;
  const int eB = blockIdx.x * 128 + (wave >> 1) * 64;

  v8f c[2][4];
#pragma unroll
  for (int i = 0; i < 2; ++i)
#pragma unroll
    for (int j = 0; j < 4; ++j) c[i][j] = {};

  for (int k = 0; k < DIM; k += 32) {
    if (k + 128 < DIM) {
      __builtin_prefetch(A + (size_t)(mB + (lane & 31)) * DIM + k + 128, 0, 0);
      __builtin_prefetch(W + (size_t)(eB + (lane & 31)) * DIM + k + 128, 0, 0);
      __builtin_prefetch(W + (size_t)(eB + 32 + (lane & 31)) * DIM + k + 128, 0, 0);
    }
    v16bf a[2], b[4];
#pragma unroll
    for (int i = 0; i < 2; ++i) a[i] = load_a(A, mB + 16 * i, DIM, k, lane);
#pragma unroll
    for (int j = 0; j < 4; ++j) b[j] = load_b(W, eB + 16 * j, DIM, k, lane);
#pragma unroll
    for (int i = 0; i < 2; ++i)
#pragma unroll
      for (int j = 0; j < 4; ++j) c[i][j] = wmma_bf16(a[i], b[j], c[i][j]);
  }
#pragma unroll
  for (int i = 0; i < 2; ++i)
#pragma unroll
    for (int j = 0; j < 4; ++j)
      store_out_tile(c[i][j], mB + 16 * i, eB + 16 * j, lane, bias, out);
}

// ---------------------------------------------------------------------------
// launch
// ---------------------------------------------------------------------------
extern "C" void kernel_launch(void* const* d_in, const int* in_sizes, int n_in,
                              void* d_out, int out_size, void* d_ws, size_t ws_size,
                              hipStream_t stream) {
  (void)in_sizes; (void)n_in; (void)out_size; (void)ws_size;
  const float* x     = (const float*)d_in[0];   // [4,2048,1024]
  const float* w_qkv = (const float*)d_in[1];   // [3072,1024]
  const float* w_out = (const float*)d_in[2];   // [1024,1024]
  const float* b_out = (const float*)d_in[3];   // [1024]
  float* out = (float*)d_out;                   // [4,2048,1024]

  // workspace layout (bf16 buffers)
  char* w = (char*)d_ws;
  size_t off = 0;
  bf16* xb    = (bf16*)(w + off); off += (size_t)MTOT * DIM * 2;            // 16 MB
  bf16* wqkvb = (bf16*)(w + off); off += (size_t)3 * DIM * DIM * 2;         //  6 MB
  bf16* wob   = (bf16*)(w + off); off += (size_t)DIM * DIM * 2;             //  2 MB
  bf16* qb    = (bf16*)(w + off); off += (size_t)BATCH * HEADS * NSEQ * HD * 2;
  bf16* kb    = (bf16*)(w + off); off += (size_t)BATCH * HEADS * NSEQ * HD * 2;
  bf16* vtb   = (bf16*)(w + off); off += (size_t)BATCH * HEADS * NSEQ * HD * 2;
  bf16* aob   = (bf16*)(w + off); off += (size_t)MTOT * DIM * 2;

  cvt_bf16_kernel<<<2048, 256, 0, stream>>>(x, xb, MTOT * DIM);
  cvt_bf16_kernel<<<2048, 256, 0, stream>>>(w_qkv, wqkvb, 3 * DIM * DIM);
  cvt_bf16_kernel<<<1024, 256, 0, stream>>>(w_out, wob, DIM * DIM);

  // QKV GEMM: M=8192, N=3072 -> 64x128 block tiles
  gemm_qkv_kernel<<<dim3(3 * DIM / 128, MTOT / 64), 128, 0, stream>>>(xb, wqkvb, qb,
                                                                      kb, vtb);
  // attention: 64 rows per block, per (b,h)
  attn_kernel<<<dim3(NSEQ / 64, BATCH * HEADS), 128, 0, stream>>>(qb, kb, vtb, aob);

  // output projection: M=8192, N=1024 -> 64x128 block tiles
  gemm_out_kernel<<<dim3(DIM / 128, MTOT / 64), 128, 0, stream>>>(aob, wob, b_out, out);
}
